// MultiHeadAttention_84507776516248
// MI455X (gfx1250) — matrix operand order repaired and checked by reading, not compiled, not run on hardware
//
#include <hip/hip_runtime.h>

typedef _Float16 half_t;
typedef __attribute__((ext_vector_type(16))) _Float16 v16h;
typedef __attribute__((ext_vector_type(8)))  float    v8f;

#define B_ 4
#define T_ 2048
#define C_ 1024
#define H_ 16
#define D_ 64

static __device__ __forceinline__ v8f wmma16x16x32(v16h a, v16h b, v8f c) {
  // D = A(16x32 f16) * B(32x16 f16) + C(16x16 f32)
  return __builtin_amdgcn_wmma_f32_16x16x32_f16(false, a, false, b, (short)0, c,
                                                false, false);
}

// Fragment loads from a precomputed per-lane pointer.
// A (16x32): lane already resolved to row + kh; 16 contiguous halves.
static __device__ __forceinline__ v16h ldA(const half_t* p) {
  v16h a;
#pragma unroll
  for (int i = 0; i < 16; ++i) a[i] = p[i];
  return a;
}
// B (32x16): lane already resolved to col + kh; 16 contiguous halves.
static __device__ __forceinline__ v16h ldB(const half_t* p) {
  v16h b;
#pragma unroll
  for (int i = 0; i < 16; ++i) b[i] = p[i];
  return b;
}

// A fragment with explicit coords (used once per tile in flash).
static __device__ __forceinline__ v16h load_a_frag(const half_t* base, int ldk,
                                                   int row0, int k0) {
  int lane = threadIdx.x & 31;
  int r = lane & 15;
  int kh = (lane < 16) ? 0 : 16;
  return ldA(base + (size_t)(row0 + r) * ldk + k0 + kh);
}

// 8 independent WMMAs for a 32x64 tile step (2 A frags x 4 B frags)
#define GEMM_STEP(A0, A1, B0, B1, B2, B3)          \
  acc[0][0] = wmma16x16x32(A0, B0, acc[0][0]);     \
  acc[0][1] = wmma16x16x32(A0, B1, acc[0][1]);     \
  acc[0][2] = wmma16x16x32(A0, B2, acc[0][2]);     \
  acc[0][3] = wmma16x16x32(A0, B3, acc[0][3]);     \
  acc[1][0] = wmma16x16x32(A1, B0, acc[1][0]);     \
  acc[1][1] = wmma16x16x32(A1, B1, acc[1][1]);     \
  acc[1][2] = wmma16x16x32(A1, B2, acc[1][2]);     \
  acc[1][3] = wmma16x16x32(A1, B3, acc[1][3]);

#define LOAD_FRAGS(A0, A1, B0, B1, B2, B3)         \
  A0 = ldA(pa);                                    \
  A1 = ldA(pa + 16 * C_);                          \
  B0 = ldB(pb);                                    \
  B1 = ldB(pb + 16 * C_);                          \
  B2 = ldB(pb + 32 * C_);                          \
  B3 = ldB(pb + 48 * C_);

// ---------------- conversion kernels ----------------

__global__ void cvt_x_kernel(const float* __restrict__ x, half_t* __restrict__ xh, int n) {
  int i = blockIdx.x * blockDim.x + threadIdx.x;
  if (i < n) xh[i] = (half_t)x[i];
}

__global__ void cvt_w_kernel(const float* __restrict__ Wq, const float* __restrict__ Wk,
                             const float* __restrict__ Wv, const float* __restrict__ Wp,
                             half_t* __restrict__ WqT, half_t* __restrict__ WkT,
                             half_t* __restrict__ WvT, half_t* __restrict__ WpT) {
  int i = blockIdx.x * blockDim.x + threadIdx.x;  // over C_*C_
  int k = i >> 10;
  int n = i & (C_ - 1);
  int o = n * C_ + k;  // transposed: Wt[n][k] = W[k][n]
  WqT[o] = (half_t)Wq[i];
  WkT[o] = (half_t)Wk[i];
  WvT[o] = (half_t)Wv[i];
  WpT[o] = (half_t)Wp[i];
}

// ---------------- fused QKV projection GEMM ----------------
// One wave computes a 32x64 tile of one of Q/K/V. Manual ping-pong software
// pipeline (X/Y fragment buffer roles) -> no rotation copies, no extra live set.
// Q,K stored [B,H,T,D] f16; V stored transposed [B,H,D,T] f16.

__global__ void __launch_bounds__(256, 1)
qkv_gemm_kernel(const half_t* __restrict__ xh,
                const half_t* __restrict__ WqT,
                const half_t* __restrict__ WkT,
                const half_t* __restrict__ WvT,
                half_t* __restrict__ Qh, half_t* __restrict__ Kh,
                half_t* __restrict__ VhT) {
  int wid = blockIdx.x * (blockDim.x >> 5) + (threadIdx.x >> 5);
  int mat = wid / (256 * 16);         // 0:Q 1:K 2:V
  int rem = wid % (256 * 16);
  int mt = rem >> 4;                  // 0..255 (32-row tiles over B*T)
  int nt = rem & 15;                  // 0..15  == head index (64-col tile)
  const half_t* Wt = (mat == 0) ? WqT : (mat == 1) ? WkT : WvT;
  int m0 = mt * 32;
  int n0 = nt * 64;

  int lane = threadIdx.x & 31;
  int l15 = lane & 15;
  int khA = (lane < 16) ? 0 : 16;
  int khB = (lane < 16) ? 0 : 16;

  const half_t* pa = xh + (size_t)(m0 + l15) * C_ + khA;
  const half_t* pb = Wt + (size_t)(n0 + l15) * C_ + khB;

  v8f acc[2][4];
#pragma unroll
  for (int u = 0; u < 2; ++u)
#pragma unroll
    for (int g = 0; g < 4; ++g)
#pragma unroll
      for (int r = 0; r < 8; ++r) acc[u][g][r] = 0.f;

  v16h xa0, xa1, xb0, xb1, xb2, xb3;
  v16h ya0, ya1, yb0, yb1, yb2, yb3;
  LOAD_FRAGS(xa0, xa1, xb0, xb1, xb2, xb3)        // k = 0

#pragma clang loop unroll(disable)
  for (int it = 0; it < 15; ++it) {
    pa += 32; pb += 32;
    LOAD_FRAGS(ya0, ya1, yb0, yb1, yb2, yb3)      // k = (2it+1)*32
    GEMM_STEP(xa0, xa1, xb0, xb1, xb2, xb3)       // k = (2it)*32
    pa += 32; pb += 32;
    LOAD_FRAGS(xa0, xa1, xb0, xb1, xb2, xb3)      // k = (2it+2)*32
    GEMM_STEP(ya0, ya1, yb0, yb1, yb2, yb3)       // k = (2it+1)*32
  }
  pa += 32; pb += 32;
  LOAD_FRAGS(ya0, ya1, yb0, yb1, yb2, yb3)        // k = 992
  GEMM_STEP(xa0, xa1, xb0, xb1, xb2, xb3)         // k = 960
  GEMM_STEP(ya0, ya1, yb0, yb1, yb2, yb3)         // k = 992

  int radd = (lane < 16) ? 0 : 8;
  int h = nt;
#pragma unroll
  for (int u = 0; u < 2; ++u) {
#pragma unroll
    for (int g = 0; g < 4; ++g) {
      int d = g * 16 + l15;
#pragma unroll
      for (int r = 0; r < 8; ++r) {
        int m = m0 + u * 16 + r + radd;
        int b = m >> 11;          // /T_
        int t = m & (T_ - 1);
        half_t val = (half_t)acc[u][g][r];
        if (mat == 0)
          Qh[(((size_t)(b * H_ + h) * T_) + t) * D_ + d] = val;
        else if (mat == 1)
          Kh[(((size_t)(b * H_ + h) * T_) + t) * D_ + d] = val;
        else
          VhT[(((size_t)(b * H_ + h) * D_) + d) * T_ + t] = val;
      }
    }
  }
}

// ---------------- flash attention ----------------
// Block = 4 waves; each wave owns one (b,h,16-row q-tile), streams 32-key tiles.
// All K and V fragment loads for a tile are issued up front from incremented
// per-lane pointers; V loads fly behind the softmax VALU chain + LDS round-trip.
// Output O written in [B,T,C] layout (row-major for the projection GEMM).

__global__ void __launch_bounds__(128, 1)
flash_attn_kernel(const half_t* __restrict__ Qh,
                  const half_t* __restrict__ Kh,
                  const half_t* __restrict__ VhT,
                  half_t* __restrict__ Oh) {
  __shared__ half_t ptile[4][16 * 32];
  int widx = threadIdx.x >> 5;
  int wid = blockIdx.x * 4 + widx;
  int b = wid / (H_ * (T_ / 16));
  int rem = wid % (H_ * (T_ / 16));
  int h = rem / (T_ / 16);
  int qt = rem % (T_ / 16);
  int m0 = qt * 16;

  const half_t* Qb = Qh + ((size_t)(b * H_ + h) * T_) * D_;
  const half_t* Kb = Kh + ((size_t)(b * H_ + h) * T_) * D_;
  const half_t* Vb = VhT + ((size_t)(b * H_ + h) * D_) * T_;
  half_t* pt = ptile[widx];

  int lane = threadIdx.x & 31;
  int l15 = lane & 15;
  int khB = (lane < 16) ? 0 : 16;
  int radd = (lane < 16) ? 0 : 8;

  v16h aq0 = load_a_frag(Qb, D_, m0, 0);
  v16h aq1 = load_a_frag(Qb, D_, m0, 32);

  // per-lane fragment base pointers, advanced by one 32-key tile each iter
  const half_t* pk = Kb + (size_t)l15 * D_ + khB;   // += 32*D_
  const half_t* pv = Vb + (size_t)l15 * T_ + khB;   // += 32

  float mrun[8], lrun[8];
  v8f oc[4];
#pragma unroll
  for (int r = 0; r < 8; ++r) { mrun[r] = -1e30f; lrun[r] = 0.f; }
#pragma unroll
  for (int g = 0; g < 4; ++g)
#pragma unroll
    for (int r = 0; r < 8; ++r) oc[g][r] = 0.f;

  const float scale = 0.125f;  // 1/sqrt(64)

#pragma clang loop unroll(disable)
  for (int kt = 0; kt < m0 + 16; kt += 32) {
    // Issue ALL loads for this tile first (K frags, then V frags).
    v16h bk00 = ldB(pk);
    v16h bk01 = ldB(pk + 32);
    v16h bk10 = ldB(pk + 16 * D_);
    v16h bk11 = ldB(pk + 16 * D_ + 32);
    v16h bv0 = ldB(pv);
    v16h bv1 = ldB(pv + 16 * T_);
    v16h bv2 = ldB(pv + 32 * T_);
    v16h bv3 = ldB(pv + 48 * T_);
    pk += 32 * D_;
    pv += 32;

    // S = Q (16x64) * K^T (64x32)
    v8f s0, s1;
#pragma unroll
    for (int r = 0; r < 8; ++r) { s0[r] = 0.f; s1[r] = 0.f; }
    s0 = wmma16x16x32(aq0, bk00, s0);
    s0 = wmma16x16x32(aq1, bk01, s0);
    s1 = wmma16x16x32(aq0, bk10, s1);
    s1 = wmma16x16x32(aq1, bk11, s1);

    // online softmax update (per C-fragment row slot r)
#pragma unroll
    for (int r = 0; r < 8; ++r) {
      int rowm = m0 + r + radd;
      float v0 = (kt + l15 <= rowm) ? s0[r] * scale : -1e30f;
      float v1 = (kt + 16 + l15 <= rowm) ? s1[r] * scale : -1e30f;
      float mx = fmaxf(v0, v1);
#pragma unroll
      for (int off = 8; off >= 1; off >>= 1) mx = fmaxf(mx, __shfl_xor(mx, off, 32));
      float mnew = fmaxf(mrun[r], mx);
      float alpha = __expf(mrun[r] - mnew);
      mrun[r] = mnew;
      float p0 = __expf(v0 - mnew);
      float p1 = __expf(v1 - mnew);
      float rs = p0 + p1;
#pragma unroll
      for (int off = 8; off >= 1; off >>= 1) rs += __shfl_xor(rs, off, 32);
      lrun[r] = lrun[r] * alpha + rs;
#pragma unroll
      for (int g = 0; g < 4; ++g) oc[g][r] *= alpha;
      int row = r + radd;
      pt[row * 32 + l15] = (half_t)p0;
      pt[row * 32 + 16 + l15] = (half_t)p1;
    }

    // O += P (16x32) * V (32x64); V fragments already resident
    v16h ap = load_a_frag(pt, 32, 0, 0);
    oc[0] = wmma16x16x32(ap, bv0, oc[0]);
    oc[1] = wmma16x16x32(ap, bv1, oc[1]);
    oc[2] = wmma16x16x32(ap, bv2, oc[2]);
    oc[3] = wmma16x16x32(ap, bv3, oc[3]);
  }

  // O /= l ; store f16 in [B,T,C] (row b*T+t, col h*64+d)
#pragma unroll
  for (int g = 0; g < 4; ++g) {
    int d = h * D_ + g * 16 + l15;
#pragma unroll
    for (int r = 0; r < 8; ++r) {
      int t = m0 + r + radd;
      float o = oc[g][r] / lrun[r];
      Oh[((size_t)(b * T_ + t)) * C_ + d] = (half_t)o;
    }
  }
}

// ---------------- output projection (+bias, f32 out) ----------------
// Oh is plain row-major [B*T, C]; identical ping-pong structure to QKV GEMM.

__global__ void __launch_bounds__(256, 1)
out_proj_kernel(const half_t* __restrict__ Oh,
                const half_t* __restrict__ WpT,
                const float* __restrict__ bp,
                float* __restrict__ out) {
  int wid = blockIdx.x * (blockDim.x >> 5) + (threadIdx.x >> 5);
  int mt = wid >> 4;   // 0..255 (32-row tiles)
  int nt = wid & 15;
  int m0 = mt * 32;
  int n0 = nt * 64;

  int lane = threadIdx.x & 31;
  int l15 = lane & 15;
  int khA = (lane < 16) ? 0 : 16;
  int khB = (lane < 16) ? 0 : 16;

  const half_t* pa = Oh + (size_t)(m0 + l15) * C_ + khA;
  const half_t* pb = WpT + (size_t)(n0 + l15) * C_ + khB;

  v8f acc[2][4];
#pragma unroll
  for (int u = 0; u < 2; ++u)
#pragma unroll
    for (int g = 0; g < 4; ++g)
#pragma unroll
      for (int r = 0; r < 8; ++r) acc[u][g][r] = 0.f;

  v16h xa0, xa1, xb0, xb1, xb2, xb3;
  v16h ya0, ya1, yb0, yb1, yb2, yb3;
  LOAD_FRAGS(xa0, xa1, xb0, xb1, xb2, xb3)

#pragma clang loop unroll(disable)
  for (int it = 0; it < 15; ++it) {
    pa += 32; pb += 32;
    LOAD_FRAGS(ya0, ya1, yb0, yb1, yb2, yb3)
    GEMM_STEP(xa0, xa1, xb0, xb1, xb2, xb3)
    pa += 32; pb += 32;
    LOAD_FRAGS(xa0, xa1, xb0, xb1, xb2, xb3)
    GEMM_STEP(ya0, ya1, yb0, yb1, yb2, yb3)
  }
  pa += 32; pb += 32;
  LOAD_FRAGS(ya0, ya1, yb0, yb1, yb2, yb3)
  GEMM_STEP(xa0, xa1, xb0, xb1, xb2, xb3)
  GEMM_STEP(ya0, ya1, yb0, yb1, yb2, yb3)

  int radd = (lane < 16) ? 0 : 8;
#pragma unroll
  for (int u = 0; u < 2; ++u) {
#pragma unroll
    for (int g = 0; g < 4; ++g) {
      int n = n0 + g * 16 + l15;
      float bias = bp[n];
#pragma unroll
      for (int r = 0; r < 8; ++r) {
        int mm = m0 + u * 16 + r + radd;
        out[(size_t)mm * C_ + n] = acc[u][g][r] + bias;
      }
    }
  }
}

// ---------------- launcher ----------------

extern "C" void kernel_launch(void* const* d_in, const int* in_sizes, int n_in,
                              void* d_out, int out_size, void* d_ws, size_t ws_size,
                              hipStream_t stream) {
  (void)in_sizes; (void)n_in; (void)out_size; (void)ws_size;
  const float* x  = (const float*)d_in[0];
  const float* Wq = (const float*)d_in[1];
  const float* Wk = (const float*)d_in[2];
  const float* Wv = (const float*)d_in[3];
  const float* Wp = (const float*)d_in[4];
  const float* bp = (const float*)d_in[5];
  float* out = (float*)d_out;

  const size_t NX = (size_t)B_ * T_ * C_;  // 8388608
  const size_t NW = (size_t)C_ * C_;       // 1048576

  half_t* xh  = (half_t*)d_ws;
  half_t* WqT = xh + NX;
  half_t* WkT = WqT + NW;
  half_t* WvT = WkT + NW;
  half_t* WpT = WvT + NW;
  half_t* Qh  = WpT + NW;
  half_t* Kh  = Qh + NX;
  half_t* VhT = Kh + NX;
  half_t* Oh  = VhT + NX;
  // total workspace: 46,137,344 halves = 88 MB

  cvt_x_kernel<<<(int)(NX / 256), 256, 0, stream>>>(x, xh, (int)NX);
  cvt_w_kernel<<<(int)(NW / 256), 256, 0, stream>>>(Wq, Wk, Wv, Wp, WqT, WkT, WvT, WpT);
  // 3 matrices * 256 m-tiles * 16 n-tiles = 12288 waves, 8 waves/block
  qkv_gemm_kernel<<<12288 / 8, 256, 0, stream>>>(xh, WqT, WkT, WvT, Qh, Kh, VhT);
  // 4*16*128 = 8192 q-tiles, 4 waves/block
  flash_attn_kernel<<<8192 / 4, 128, 0, stream>>>(Qh, Kh, VhT, Oh);
  // 256*16 = 4096 waves, 8 waves/block
  out_proj_kernel<<<4096 / 8, 256, 0, stream>>>(Oh, WpT, bp, out);
}
